// SATSolver_37864431681665
// MI455X (gfx1250) — matrix operand order repaired and checked by reading, not compiled, not run on hardware
//
#include <hip/hip_runtime.h>
#include <math.h>

// ---------------- problem constants ----------------
constexpr int kB = 8, kC = 2048, kL = 3, kV = 512, kH = 256;
constexpr int kHeads = 8, kHD = 32, kIters = 8;
constexpr int kBC = kB * kC;   // 16384
constexpr int kBV = kB * kV;   // 4096

// ---------------- WMMA types (CDNA5, wave32) ----------------
typedef __attribute__((ext_vector_type(16))) __bf16 bf16x16;
typedef __attribute__((ext_vector_type(8)))  float  f32x8;

__device__ inline f32x8 wmma_bf16(bf16x16 a, bf16x16 b, f32x8 c) {
  // D = A(16x32 bf16) * B(32x16 bf16) + C(16x16 f32)
  return __builtin_amdgcn_wmma_f32_16x16x32_bf16(false, a, false, b, (short)0, c,
                                                 false, false);
}

__device__ inline float gelu_f(float x) {
  return 0.5f * x * (1.0f + erff(x * 0.70710678118654752f));
}

__device__ inline f32x8 zero8() {
  f32x8 z;
#pragma unroll
  for (int e = 0; e < 8; ++e) z[e] = 0.0f;
  return z;
}

// =====================================================================
// Generic GEMM: Out[m,n] = act( (rowScale[m]*A[m,:]) . W[n,:] + bias[n] )
// A: MxK f32 row-major, W: NxK f32 row-major (torch-style), Out: MxN f32.
// grid = (M/16, N/64), block = 32 (one wave). 4 WMMAs per 32-wide K step.
// A-operand k order used here (contiguous, same as B): lane l (g=l/16, m=l%16):
//   elems 0..7  -> K = g*16 + j       elems 8..15 -> K = g*16 + 8 + j
// B-operand (32x16), same k order: lane l: elem e -> B[g*16+e, l%16]
// D layout: lane l, elem v -> D[v + 8*g, l%16]
// =====================================================================
__global__ void k_gemm(const float* __restrict__ A, const float* __restrict__ W,
                       const float* __restrict__ bias,
                       const float* __restrict__ rowScale,
                       float* __restrict__ Out, int M, int N, int Kd,
                       int fuseGelu) {
  const int lane = threadIdx.x & 31;
  const int g = lane >> 4, lr = lane & 15;
  const int m0 = blockIdx.x * 16, n0 = blockIdx.y * 64;
  const int row = m0 + lr;
  const float rs = rowScale ? rowScale[row] : 1.0f;
  f32x8 acc[4];
#pragma unroll
  for (int t = 0; t < 4; ++t) acc[t] = zero8();
  const float* ap = A + (size_t)row * Kd;
  for (int kb = 0; kb < Kd; kb += 32) {
    if (kb + 32 < Kd) {
      // warm L0/WGP$ for the next K-step (global_prefetch_b8, no counter cost)
      __builtin_prefetch(ap + kb + 32 + g * 16, 0, 1);
    }
    bf16x16 a;
#pragma unroll
    for (int j = 0; j < 8; ++j) {
      a[j]     = (__bf16)(ap[kb + g * 16 + j] * rs);
      a[j + 8] = (__bf16)(ap[kb + g * 16 + 8 + j] * rs);
    }
#pragma unroll
    for (int t = 0; t < 4; ++t) {
      const float* wp = W + (size_t)(n0 + t * 16 + lr) * Kd + kb + g * 16;
      if (kb + 32 < Kd) __builtin_prefetch(wp + 32, 0, 1);
      bf16x16 bm;
#pragma unroll
      for (int e = 0; e < 16; ++e) bm[e] = (__bf16)wp[e];
      acc[t] = wmma_bf16(a, bm, acc[t]);
    }
  }
#pragma unroll
  for (int t = 0; t < 4; ++t) {
    const int n = n0 + t * 16 + lr;
    const float bv = bias ? bias[n] : 0.0f;
#pragma unroll
    for (int e = 0; e < 8; ++e) {
      float x = acc[t][e] + bv;
      if (fuseGelu) x = gelu_f(x);
      Out[(size_t)(m0 + e + 8 * g) * N + n] = x;
    }
  }
}

// =====================================================================
// Flash attention: one wave handles (b, h, 16 query rows), online softmax
// over C keys in 32-key tiles. S tile via 2 WMMAs (K = HD = 32 exactly),
// O accumulation via 2 WMMAs (P 16x32 @ V 32x32 -> two 16x16 dim tiles).
// =====================================================================
__global__ void k_flash(const float* __restrict__ Q, const float* __restrict__ Kmat,
                        const float* __restrict__ Vmat, float* __restrict__ O) {
  const int lane = threadIdx.x & 31;
  const int g = lane >> 4, lr = lane & 15;
  const int r0 = blockIdx.x * 16;   // query row block
  const int h = blockIdx.y, b = blockIdx.z;
  const int d0 = h * kHD;
  __shared__ __bf16 pl[16 * 32];    // P tile (rows x keys)
  __shared__ __bf16 vl[32 * 32];    // V tile (keys x dims)

  bf16x16 qa;
  {
    const float sc = 0.17677669529663687f;  // 1/sqrt(HD), folded into Q
    const float* qp = Q + (size_t)(b * kV + r0 + lr) * kH + d0;
#pragma unroll
    for (int j = 0; j < 8; ++j) {
      qa[j]     = (__bf16)(qp[g * 16 + j] * sc);
      qa[j + 8] = (__bf16)(qp[g * 16 + 8 + j] * sc);
    }
  }
  float mrun[8], lrun[8];
  f32x8 o0 = zero8(), o1 = zero8();
#pragma unroll
  for (int e = 0; e < 8; ++e) { mrun[e] = -3.0e38f; lrun[e] = 0.0f; }

  for (int kt = 0; kt < kC; kt += 32) {
    if (kt + 32 < kC) {
      // prefetch next key tile rows of K and V while this tile computes
      __builtin_prefetch(Kmat + (size_t)(b * kC + kt + 32 + lr) * kH + d0, 0, 1);
      __builtin_prefetch(Vmat + (size_t)(b * kC + kt + 32 + lane) * kH + d0, 0, 1);
    }
    // K^T operands for two 16-key groups (contraction dim = HD)
    bf16x16 kb0, kb1;
    const float* kp0 = Kmat + (size_t)(b * kC + kt + lr) * kH + d0 + g * 16;
    const float* kp1 = kp0 + (size_t)16 * kH;
#pragma unroll
    for (int e = 0; e < 16; ++e) { kb0[e] = (__bf16)kp0[e]; kb1[e] = (__bf16)kp1[e]; }
    f32x8 s0 = wmma_bf16(qa, kb0, zero8());
    f32x8 s1 = wmma_bf16(qa, kb1, zero8());
    // stage V tile (key kt+lane, 32 dims) in LDS
    {
      const float* vp = Vmat + (size_t)(b * kC + kt + lane) * kH + d0;
#pragma unroll
      for (int e = 0; e < 32; ++e) vl[lane * 32 + e] = (__bf16)vp[e];
    }
    // online softmax per row (rows of group g live in lanes of group g)
    float alpha[8];
#pragma unroll
    for (int e = 0; e < 8; ++e) {
      float t = fmaxf(s0[e], s1[e]);
      t = fmaxf(t, __shfl_xor(t, 1, 32));
      t = fmaxf(t, __shfl_xor(t, 2, 32));
      t = fmaxf(t, __shfl_xor(t, 4, 32));
      t = fmaxf(t, __shfl_xor(t, 8, 32));
      const float mnew = fmaxf(mrun[e], t);
      alpha[e] = __expf(mrun[e] - mnew);
      mrun[e] = mnew;
      const float p0 = __expf(s0[e] - mnew);
      const float p1 = __expf(s1[e] - mnew);
      float rsum = p0 + p1;
      rsum += __shfl_xor(rsum, 1, 32);
      rsum += __shfl_xor(rsum, 2, 32);
      rsum += __shfl_xor(rsum, 4, 32);
      rsum += __shfl_xor(rsum, 8, 32);
      lrun[e] = lrun[e] * alpha[e] + rsum;
      // D layout -> LDS row-major [row][key]
      pl[(e + 8 * g) * 32 + lr]      = (__bf16)p0;
      pl[(e + 8 * g) * 32 + 16 + lr] = (__bf16)p1;
      o0[e] *= alpha[e];
      o1[e] *= alpha[e];
    }
    __syncthreads();  // single-wave workgroup: barrier->NOP, DS wait only
    // reload P in A-operand layout, V tiles in B-operand layout
    bf16x16 pa, vb0, vb1;
#pragma unroll
    for (int j = 0; j < 8; ++j) {
      pa[j]     = pl[lr * 32 + g * 16 + j];
      pa[j + 8] = pl[lr * 32 + g * 16 + 8 + j];
    }
#pragma unroll
    for (int e = 0; e < 16; ++e) {
      vb0[e] = vl[(g * 16 + e) * 32 + lr];
      vb1[e] = vl[(g * 16 + e) * 32 + 16 + lr];
    }
    o0 = wmma_bf16(pa, vb0, o0);
    o1 = wmma_bf16(pa, vb1, o1);
    __syncthreads();
  }
#pragma unroll
  for (int e = 0; e < 8; ++e) {
    const float inv = 1.0f / lrun[e];
    float* op = O + (size_t)(b * kV + r0 + e + 8 * g) * kH + d0;
    op[lr]      = o0[e] * inv;
    op[16 + lr] = o1[e] * inv;
  }
}

// ---------------- embedding gather + mean over L ----------------
__global__ void k_embed(const int* __restrict__ formula,
                        const float* __restrict__ emb, float* __restrict__ X) {
  const int idx = blockIdx.x;   // b*C + c
  const int t = threadIdx.x;    // 0..H-1
  const int f0 = formula[idx * kL + 0];
  const int f1 = formula[idx * kL + 1];
  const int f2 = formula[idx * kL + 2];
  const float s = emb[(size_t)f0 * kH + t] + emb[(size_t)f1 * kH + t] +
                  emb[(size_t)f2 * kH + t];
  X[(size_t)idx * kH + t] = s * (1.0f / 3.0f);
}

// ---------------- LayerNorm(512) + exact GELU, in place ----------------
__global__ void k_ln_gelu(float* __restrict__ hid, const float* __restrict__ gam,
                          const float* __restrict__ bet) {
  const int t = threadIdx.x, lane = t & 31, wv = t >> 5;
  const int row = blockIdx.x * 8 + wv;
  float* rp = hid + (size_t)row * (2 * kH);
  float x[16];
  float s = 0.0f;
#pragma unroll
  for (int j = 0; j < 16; ++j) { x[j] = rp[lane + 32 * j]; s += x[j]; }
  for (int m = 16; m >= 1; m >>= 1) s += __shfl_xor(s, m, 32);
  const float mean = s * (1.0f / 512.0f);
  float vv = 0.0f;
#pragma unroll
  for (int j = 0; j < 16; ++j) { const float d = x[j] - mean; vv += d * d; }
  for (int m = 16; m >= 1; m >>= 1) vv += __shfl_xor(vv, m, 32);
  const float rstd = rsqrtf(vv * (1.0f / 512.0f) + 1e-5f);
#pragma unroll
  for (int j = 0; j < 16; ++j) {
    const int col = lane + 32 * j;
    const float y = (x[j] - mean) * rstd * gam[col] + bet[col];
    rp[col] = gelu_f(y);
  }
}

// ---------------- clause sat gate: 1 - 0.9*sat ----------------
__global__ void k_sat(const int* __restrict__ formula, const int* __restrict__ assign,
                      float* __restrict__ satscale) {
  const int idx = blockIdx.x * 256 + threadIdx.x;
  if (idx >= kBC) return;
  const int b = idx / kC;
  bool sat = false;
#pragma unroll
  for (int l = 0; l < kL; ++l) {
    const int f = formula[idx * kL + l];
    const int w = f >> 1, p = f & 1;
    const int a = assign[b * kV + w];
    sat |= (a == p);  // a==p (p<2) implies a!=2
  }
  satscale[idx] = sat ? 0.1f : 1.0f;
}

// ---------------- scores = h1 . vs_w2 + b ----------------
__global__ void k_score(const float* __restrict__ h1, const float* __restrict__ w2,
                        const float* __restrict__ b2, float* __restrict__ scores) {
  const int t = threadIdx.x, lane = t & 31, wv = t >> 5;
  const int row = blockIdx.x * 8 + wv;
  const float* rp = h1 + (size_t)row * kH;
  float s = 0.0f;
#pragma unroll
  for (int j = 0; j < 8; ++j) s += rp[lane + 32 * j] * w2[lane + 32 * j];
  for (int m = 16; m >= 1; m >>= 1) s += __shfl_xor(s, m, 32);
  if (lane == 0) scores[row] = s + b2[0];
}

// ---------------- masked argmax (first max, jnp semantics) ----------------
__global__ void k_argmax(const float* __restrict__ scores,
                         const int* __restrict__ assign, int* __restrict__ sel) {
  const int b = blockIdx.x, t = threadIdx.x;
  __shared__ float sv[256];
  __shared__ int si[256];
  float best = -3.4e38f;
  int bi = kV;
  for (int v = t; v < kV; v += 256) {
    float s = scores[b * kV + v];
    if (assign[b * kV + v] != 2) s -= 1.0e9f;
    if (s > best || (s == best && v < bi)) { best = s; bi = v; }
  }
  sv[t] = best; si[t] = bi;
  __syncthreads();
  for (int off = 128; off > 0; off >>= 1) {
    if (t < off) {
      if (sv[t + off] > sv[t] || (sv[t + off] == sv[t] && si[t + off] < si[t])) {
        sv[t] = sv[t + off]; si[t] = si[t + off];
      }
    }
    __syncthreads();
  }
  if (t == 0) sel[b] = si[0];
}

// ---------------- policy head on selected state ----------------
__global__ void k_policy(const float* __restrict__ vsb, const int* __restrict__ sel,
                         const float* __restrict__ w1, const float* __restrict__ b1,
                         const float* __restrict__ w2, const float* __restrict__ b2,
                         int* __restrict__ newa) {
  const int b = blockIdx.x, t = threadIdx.x;
  __shared__ float xx[kH], hh[kH];
  xx[t] = vsb[(size_t)(b * kV + sel[b]) * kH + t];
  __syncthreads();
  float acc = b1[t];
  for (int k2 = 0; k2 < kH; ++k2) acc += w1[t * kH + k2] * xx[k2];
  hh[t] = gelu_f(acc);
  __syncthreads();
  if (t == 0) {
    float l0 = b2[0], l1 = b2[1];
    for (int k2 = 0; k2 < kH; ++k2) { l0 += w2[k2] * hh[k2]; l1 += w2[kH + k2] * hh[k2]; }
    newa[b] = (l1 > l0) ? 1 : 0;  // argmax(softmax) with first-index ties
  }
}

// flags: [0]=done, [1]=any_conf, [2]=all_assigned
__global__ void k_reset(int* __restrict__ flags) {
  if (threadIdx.x == 0) { flags[1] = 0; flags[2] = 1; }
}

__global__ void k_conflict(const int* __restrict__ formula, const int* __restrict__ assign,
                           const int* __restrict__ sel, const int* __restrict__ newa,
                           int* __restrict__ flags) {
  const int idx = blockIdx.x * 256 + threadIdx.x;
  if (idx >= kBC) return;
  const int b = idx / kC;
  bool sat = false, allass = true;
#pragma unroll
  for (int l = 0; l < kL; ++l) {
    const int f = formula[idx * kL + l];
    const int w = f >> 1, p = f & 1;
    const int a = (w == sel[b]) ? newa[b] : assign[b * kV + w];
    sat |= (a == p);
    allass &= (a != 2);
  }
  if (allass && !sat) atomicOr(&flags[1], 1);
}

__global__ void k_assigned(const int* __restrict__ assign, const int* __restrict__ sel,
                           const int* __restrict__ newa, int* __restrict__ flags) {
  const int idx = blockIdx.x * 256 + threadIdx.x;
  if (idx >= kBV) return;
  const int b = idx / kV, v = idx % kV;
  const int a = (v == sel[b]) ? newa[b] : assign[idx];
  if (a == 2) atomicAnd(&flags[2], 0);
}

__global__ void k_commit(const int* __restrict__ flags, const float* __restrict__ vsb,
                         float* __restrict__ varst, int* __restrict__ assign,
                         const int* __restrict__ sel, const int* __restrict__ newa) {
  if (flags[0]) return;  // carry frozen once done
  const size_t idx = (size_t)blockIdx.x * 256 + threadIdx.x;
  varst[idx] = vsb[idx];
  if ((idx & (size_t)(kH - 1)) == 0) {
    const int bv = (int)(idx / kH);
    const int b = bv / kV, v = bv % kV;
    if (v == sel[b]) assign[bv] = flags[1] ? 2 : newa[b];
  }
}

__global__ void k_done(int* __restrict__ flags) {
  if (threadIdx.x == 0) {
    if (!flags[0] && flags[2] && !flags[1]) flags[0] = 1;
  }
}

__global__ void k_init(float* __restrict__ varst, int* __restrict__ assign,
                       int* __restrict__ flags) {
  const size_t idx = (size_t)blockIdx.x * 256 + threadIdx.x;
  varst[idx] = 0.0f;
  if (idx < (size_t)kBV) assign[idx] = 2;
  if (idx == 0) flags[0] = 0;
}

__global__ void k_meanvs(const float* __restrict__ varst, float* __restrict__ mv) {
  const int b = blockIdx.x, t = threadIdx.x;
  float s = 0.0f;
  for (int v = 0; v < kV; ++v) s += varst[(size_t)(b * kV + v) * kH + t];
  mv[b * kH + t] = s * (1.0f / kV);
}

__global__ void k_final(const float* __restrict__ mv, const float* __restrict__ w1,
                        const float* __restrict__ b1, const float* __restrict__ w2,
                        const float* __restrict__ b2, float* __restrict__ out) {
  const int b = blockIdx.x, t = threadIdx.x;
  __shared__ float xx[kH], hh[kH];
  xx[t] = mv[b * kH + t];
  __syncthreads();
  float acc = b1[t];
  for (int k2 = 0; k2 < kH; ++k2) acc += w1[t * kH + k2] * xx[k2];
  hh[t] = gelu_f(acc);
  __syncthreads();
  if (t == 0) {
    float s = b2[0];
    for (int k2 = 0; k2 < kH; ++k2) s += w2[k2] * hh[k2];
    out[b] = 1.0f / (1.0f + __expf(-s));
  }
}

__global__ void k_outassign(const int* __restrict__ assign, float* __restrict__ out) {
  const int idx = blockIdx.x * 256 + threadIdx.x;
  if (idx < kBV) out[kB + idx] = (float)assign[idx];
}

// =====================================================================
extern "C" void kernel_launch(void* const* d_in, const int* in_sizes, int n_in,
                              void* d_out, int out_size, void* d_ws, size_t ws_size,
                              hipStream_t stream) {
  (void)in_sizes; (void)n_in; (void)out_size; (void)ws_size;
  const int*   formula    = (const int*)d_in[0];
  const float* lit_emb    = (const float*)d_in[1];
  const float* ce_w1      = (const float*)d_in[2];
  const float* ce_b1      = (const float*)d_in[3];
  const float* ce_g       = (const float*)d_in[4];
  const float* ce_beta    = (const float*)d_in[5];
  const float* ce_w2      = (const float*)d_in[6];
  const float* ce_b2      = (const float*)d_in[7];
  const float* attn_in_w  = (const float*)d_in[8];
  const float* attn_in_b  = (const float*)d_in[9];
  const float* attn_out_w = (const float*)d_in[10];
  const float* attn_out_b = (const float*)d_in[11];
  const float* vs_w1      = (const float*)d_in[12];
  const float* vs_b1      = (const float*)d_in[13];
  const float* vs_w2      = (const float*)d_in[14];
  const float* vs_b2      = (const float*)d_in[15];
  const float* ap_w1      = (const float*)d_in[16];
  const float* ap_b1      = (const float*)d_in[17];
  const float* ap_w2      = (const float*)d_in[18];
  const float* ap_b2      = (const float*)d_in[19];
  const float* sp_w1      = (const float*)d_in[20];
  const float* sp_b1      = (const float*)d_in[21];
  const float* sp_w2      = (const float*)d_in[22];
  const float* sp_b2      = (const float*)d_in[23];
  float* out = (float*)d_out;

  // workspace carve-out (everything fp32; ~84 MB, L2-resident on MI455X)
  char* wsp = (char*)d_ws;
  auto alloc = [&](size_t bytes) -> void* {
    void* p = (void*)wsp;
    wsp += (bytes + 255) & ~(size_t)255;
    return p;
  };
  float* X     = (float*)alloc((size_t)kBC * kH * 4);      // lit-mean; reused as K
  float* hid   = (float*)alloc((size_t)kBC * 2 * kH * 4);  // ce hidden; reused as V
  float* base  = (float*)alloc((size_t)kBC * kH * 4);
  float* Qb    = (float*)alloc((size_t)kBV * kH * 4);
  float* attnO = (float*)alloc((size_t)kBV * kH * 4);
  float* vsb   = (float*)alloc((size_t)kBV * kH * 4);
  float* h1b   = (float*)alloc((size_t)kBV * kH * 4);
  float* varst = (float*)alloc((size_t)kBV * kH * 4);
  float* satsc = (float*)alloc((size_t)kBC * 4);
  float* scores= (float*)alloc((size_t)kBV * 4);
  float* mv    = (float*)alloc((size_t)kB * kH * 4);
  int* assign  = (int*)alloc((size_t)kBV * 4);
  int* sel     = (int*)alloc(kB * 4);
  int* newa    = (int*)alloc(kB * 4);
  int* flags   = (int*)alloc(64);
  float* Kb = X;    // X is dead after the clause encoder
  float* Vb = hid;  // hid is dead after the clause encoder

  // init carries
  k_init<<<dim3((kBV * kH) / 256), 256, 0, stream>>>(varst, assign, flags);

  // clause encoder: gather+mean -> GEMM(H->2H) -> LN+GELU -> GEMM(2H->H)
  k_embed<<<dim3(kBC), kH, 0, stream>>>(formula, lit_emb, X);
  k_gemm<<<dim3(kBC / 16, (2 * kH) / 64), 32, 0, stream>>>(
      X, ce_w1, ce_b1, nullptr, hid, kBC, 2 * kH, kH, 0);
  k_ln_gelu<<<dim3(kBC / 8), 256, 0, stream>>>(hid, ce_g, ce_beta);
  k_gemm<<<dim3(kBC / 16, kH / 64), 32, 0, stream>>>(
      hid, ce_w2, ce_b2, nullptr, base, kBC, kH, 2 * kH, 0);

  for (int it = 0; it < kIters; ++it) {
    k_sat<<<dim3((kBC + 255) / 256), 256, 0, stream>>>(formula, assign, satsc);
    // Q from var_states; K/V from sat-gated clause states (gate fused as rowScale)
    k_gemm<<<dim3(kBV / 16, kH / 64), 32, 0, stream>>>(
        varst, attn_in_w, attn_in_b, nullptr, Qb, kBV, kH, kH, 0);
    k_gemm<<<dim3(kBC / 16, kH / 64), 32, 0, stream>>>(
        base, attn_in_w + kH * kH, attn_in_b + kH, satsc, Kb, kBC, kH, kH, 0);
    k_gemm<<<dim3(kBC / 16, kH / 64), 32, 0, stream>>>(
        base, attn_in_w + 2 * kH * kH, attn_in_b + 2 * kH, satsc, Vb, kBC, kH, kH, 0);
    k_flash<<<dim3(kV / 16, kHeads, kB), 32, 0, stream>>>(Qb, Kb, Vb, attnO);
    k_gemm<<<dim3(kBV / 16, kH / 64), 32, 0, stream>>>(
        attnO, attn_out_w, attn_out_b, nullptr, vsb, kBV, kH, kH, 0);
    // variable-select MLP
    k_gemm<<<dim3(kBV / 16, kH / 64), 32, 0, stream>>>(
        vsb, vs_w1, vs_b1, nullptr, h1b, kBV, kH, kH, 1);
    k_score<<<dim3(kBV / 8), 256, 0, stream>>>(h1b, vs_w2, vs_b2, scores);
    k_argmax<<<dim3(kB), 256, 0, stream>>>(scores, assign, sel);
    k_policy<<<dim3(kB), kH, 0, stream>>>(vsb, sel, ap_w1, ap_b1, ap_w2, ap_b2, newa);
    k_reset<<<dim3(1), 32, 0, stream>>>(flags);
    k_conflict<<<dim3((kBC + 255) / 256), 256, 0, stream>>>(formula, assign, sel, newa, flags);
    k_assigned<<<dim3((kBV + 255) / 256), 256, 0, stream>>>(assign, sel, newa, flags);
    k_commit<<<dim3((kBV * kH) / 256), 256, 0, stream>>>(flags, vsb, varst, assign, sel, newa);
    k_done<<<dim3(1), 32, 0, stream>>>(flags);
  }

  // outputs: is_sat (B) then assignments (B*V) as floats
  k_meanvs<<<dim3(kB), kH, 0, stream>>>(varst, mv);
  k_final<<<dim3(kB), kH, 0, stream>>>(mv, sp_w1, sp_b1, sp_w2, sp_b2, out);
  k_outassign<<<dim3((kBV + 255) / 256), 256, 0, stream>>>(assign, out);
}
